// GenericLossFirstPart_Sim_40647570490025
// MI455X (gfx1250) — compile-verified, matrix operand order checked by reading, not run on hardware
//
#include <hip/hip_runtime.h>
#include <hip/hip_bf16.h>

typedef __bf16 bf16;
typedef __attribute__((ext_vector_type(16))) __bf16 v16bf;
typedef __attribute__((ext_vector_type(8)))  float  v8f;

#define B_M 128
#define B_N 128
#define KDIM 256
#define LDS_K (KDIM + 8)   // ushort units; +16B/row pad -> 528B pitch, conflict-free b128 gathers

__device__ __forceinline__ unsigned pack2bf(float x, float y) {
    unsigned ux = __float_as_uint(x), uy = __float_as_uint(y);
    ux = (ux + 0x7fffu + ((ux >> 16) & 1u)) >> 16;   // RNE f32 -> bf16
    uy = (uy + 0x7fffu + ((uy >> 16) & 1u)) >> 16;
    return ux | (uy << 16);
}

__device__ __forceinline__ float waveReduceSum(float v) {
    #pragma unroll
    for (int off = 16; off; off >>= 1) v += __shfl_xor(v, off, 32);
    return v;
}

// ---- weights: f32 -> bf16 (padded to 1024 rows) + row norms ----
__global__ __launch_bounds__(256) void prep_weights(
    const float* __restrict__ W, unsigned short* __restrict__ Bbf,
    float* __restrict__ wn)
{
    __shared__ float red[8];
    const int o = blockIdx.x;     // 0..1023
    const int t = threadIdx.x;    // 0..255 == K index
    float w = 0.f;
    if (o < 1000) w = W[o * 256 + t];
    unsigned ux = __float_as_uint(w);
    ux = (ux + 0x7fffu + ((ux >> 16) & 1u)) >> 16;
    Bbf[o * 256 + t] = (unsigned short)ux;
    float s = waveReduceSum(w * w);
    const int wave = t >> 5, lane = t & 31;
    if (lane == 0) red[wave] = s;
    __syncthreads();
    if (t == 0) {
        float tot = 0.f;
        #pragma unroll
        for (int i = 0; i < 8; ++i) tot += red[i];
        wn[o] = (o < 1000) ? sqrtf(tot) : 1.0f;
    }
}

// ---- feature row norms: one wave per row ----
__global__ __launch_bounds__(256) void feat_norms(
    const float* __restrict__ F, float* __restrict__ fn)
{
    const int wave = threadIdx.x >> 5, lane = threadIdx.x & 31;
    const int row = blockIdx.x * 8 + wave;
    const float4* Fr = (const float4*)(F + (size_t)row * KDIM);
    float4 a = Fr[lane], b = Fr[lane + 32];
    float s = a.x*a.x + a.y*a.y + a.z*a.z + a.w*a.w
            + b.x*b.x + b.y*b.y + b.z*b.z + b.w*b.w;
    s = waveReduceSum(s);
    if (lane == 0) fn[row] = sqrtf(s);
}

// ---- fused cosine GEMM: 128x128 tile, bf16 WMMA, full-K LDS residency ----
__global__ __launch_bounds__(256) void cosine_gemm(
    const float* __restrict__ F, const unsigned short* __restrict__ Bbf,
    const float* __restrict__ fn, const float* __restrict__ wnorm,
    const int* __restrict__ alpha_p, float* __restrict__ out)
{
    __shared__ unsigned short sA[B_M * LDS_K];  // 67584 B
    __shared__ unsigned short sB[B_N * LDS_K];  // 67584 B

    const int t = threadIdx.x;
    const int Mbase = blockIdx.y * B_M;
    const int Nbase = blockIdx.x * B_N;

    // A tile: 128 rows x 256 f32 -> bf16. 8192 float4 / 256 thr = 32 each, coalesced.
    {
        const float4* Ag = (const float4*)F;
        #pragma unroll
        for (int i = 0; i < 32; ++i) {
            int f = i * 256 + t;
            int row = f >> 6;          // 64 float4 per row
            int c4  = f & 63;
            float4 v = Ag[(size_t)(Mbase + row) * 64 + c4];
            uint2 p;
            p.x = pack2bf(v.x, v.y);
            p.y = pack2bf(v.z, v.w);
            *(uint2*)&sA[row * LDS_K + c4 * 4] = p;
        }
    }
    // B tile: 128 rows x 256 bf16 copy. 4096 uint4 / 256 thr = 16 each.
    {
        const uint4* Bg = (const uint4*)Bbf;
        #pragma unroll
        for (int i = 0; i < 16; ++i) {
            int f = i * 256 + t;
            int row = f >> 5;          // 32 uint4 per row
            int c   = f & 31;
            uint4 v = Bg[(size_t)(Nbase + row) * 32 + c];
            *(uint4*)&sB[row * LDS_K + c * 8] = v;
        }
    }
    __syncthreads();

    const int lane = t & 31;
    const int half = (t >> 4) & 1;
    const int l16  = lane & 15;
    const int wave = t >> 5;
    const int wm  = wave & 3;    // M sub-tile: wm*32
    const int wn_ = wave >> 2;   // N sub-tile: wn_*64

    v8f acc[2][4];
    {
        v8f z = {};
        #pragma unroll
        for (int i = 0; i < 2; ++i)
            #pragma unroll
            for (int j = 0; j < 4; ++j) acc[i][j] = z;
    }

    union Frag { uint4 u[2]; v16bf v; };

    #pragma unroll
    for (int ks = 0; ks < 8; ++ks) {
        const int k0 = ks * 32 + half * 8;   // per-lane K origin (bf16 A/B layout)
        Frag a[2], b[4];
        #pragma unroll
        for (int mt = 0; mt < 2; ++mt) {
            const unsigned short* p = &sA[(wm * 32 + mt * 16 + l16) * LDS_K + k0];
            a[mt].u[0] = *(const uint4*)p;          // K = k0..k0+7
            a[mt].u[1] = *(const uint4*)(p + 16);   // K = k0+16..k0+23
        }
        #pragma unroll
        for (int nt = 0; nt < 4; ++nt) {
            const unsigned short* p = &sB[(wn_ * 64 + nt * 16 + l16) * LDS_K + k0];
            b[nt].u[0] = *(const uint4*)p;
            b[nt].u[1] = *(const uint4*)(p + 16);
        }
        #pragma unroll
        for (int mt = 0; mt < 2; ++mt)
            #pragma unroll
            for (int nt = 0; nt < 4; ++nt)
                acc[mt][nt] = __builtin_amdgcn_wmma_f32_16x16x32_bf16(
                    false, a[mt].v, false, b[nt].v,
                    (short)0, acc[mt][nt], false, false);
    }

    // fused epilogue: -alpha * dot / max(|f|*|w|, eps)
    const float nalpha = -(float)(*alpha_p);
    #pragma unroll
    for (int mt = 0; mt < 2; ++mt) {
        #pragma unroll
        for (int nt = 0; nt < 4; ++nt) {
            const int colg = Nbase + wn_ * 64 + nt * 16 + l16;
            if (colg >= 1000) continue;
            const float w = wnorm[colg];
            #pragma unroll
            for (int r = 0; r < 8; ++r) {
                const int rowg = Mbase + wm * 32 + mt * 16 + half * 8 + r;
                const float fno = fn[rowg];
                const float denom = fmaxf(fno * w, 1e-8f);
                out[(size_t)rowg * 1000 + colg] = nalpha * acc[mt][nt][r] / denom;
            }
        }
    }
}

extern "C" void kernel_launch(void* const* d_in, const int* in_sizes, int n_in,
                              void* d_out, int out_size, void* d_ws, size_t ws_size,
                              hipStream_t stream) {
    const float* F     = (const float*)d_in[0];   // 65536 x 256
    const float* W     = (const float*)d_in[1];   // 1000 x 256
    const int*   alpha = (const int*)d_in[2];     // scalar
    float* out = (float*)d_out;

    char* ws = (char*)d_ws;
    float* fn = (float*)ws;                                   // 65536 f32
    float* wn = (float*)(ws + 65536 * sizeof(float));         // 1024 f32
    unsigned short* Bbf =
        (unsigned short*)(ws + 65536 * sizeof(float) + 1024 * sizeof(float)); // 1024x256 bf16

    prep_weights<<<1024, 256, 0, stream>>>(W, Bbf, wn);
    feat_norms  <<<8192, 256, 0, stream>>>(F, fn);
    cosine_gemm <<<dim3(8, 512), 256, 0, stream>>>(F, Bbf, fn, wn, alpha, out);
}
